// SeaLiceGLKAN_49143015801405
// MI455X (gfx1250) — matrix operand order repaired and checked by reading, not compiled, not run on hardware
//
#include <hip/hip_runtime.h>
#include <hip/hip_bf16.h>
#include <stdint.h>

#define BB 2
#define TT 4
#define NN 10000
#define FF 16
#define HH 64
#define EE 160000
#define NBASIS 8
#define KHOP 3
#define BN (BB*NN)

typedef __attribute__((ext_vector_type(16))) _Float16 v16h;
typedef __attribute__((ext_vector_type(8)))  float    v8f;
typedef unsigned int u32x4 __attribute__((ext_vector_type(4)));
typedef int          i32x4 __attribute__((ext_vector_type(4)));
typedef int          i32x8 __attribute__((ext_vector_type(8)));

// fast hardware reciprocal (v_rcp_f32, single TRANS op)
__device__ __forceinline__ float frcp(float x) { return __builtin_amdgcn_rcpf(x); }
__device__ __forceinline__ float fsigmoid(float x) { return frcp(1.f + __expf(-x)); }

// ---- monotone float<->uint encoding for atomicMax-based segment_max ----
__device__ __forceinline__ unsigned fenc(float f) {
  unsigned u = __float_as_uint(f);
  return (u & 0x80000000u) ? ~u : (u | 0x80000000u);
}
__device__ __forceinline__ float fdec(unsigned u) {
  return (u & 0x80000000u) ? __uint_as_float(u & 0x7fffffffu) : __uint_as_float(~u);
}
#define ENC_NEG_INF 0x007FFFFFu

// A-fragment K index for 16-bit A (16x32): element j of lane-half `half`
__device__ __forceinline__ int a_k(int j, int half) {
  return half * 16 + j;
}

// ---- Tensor Data Mover: 1-D copy of (n8 * 8) bytes global -> LDS ----------
__device__ __forceinline__ void tdm_load_weights(_Float16* lds, const _Float16* g, int n8) {
  if (threadIdx.x == 0) {
    unsigned lo = (unsigned)(uintptr_t)lds;            // generic addr low 32 = LDS offset
    unsigned long long ga = (unsigned long long)(uintptr_t)g;
    u32x4 g0;
    g0[0] = 1u;                                        // count=1, user descriptor
    g0[1] = lo;                                        // lds_addr (bytes)
    g0[2] = (unsigned)ga;                              // global_addr[31:0]
    g0[3] = (unsigned)((ga >> 32) & 0x01ffffffull) | (2u << 30);  // addr[56:32] | type=2
    i32x8 g1;
    g1[0] = (int)(3u << 16);                           // workgroup_mask=0, data_size=8B
    g1[1] = (int)(((unsigned)n8 & 0xffffu) << 16);     // tensor_dim0[15:0]
    g1[2] = (int)((((unsigned)n8 >> 16) & 0xffffu) | (1u << 16)); // td0[31:16] | td1=1
    g1[3] = (int)(((unsigned)n8 & 0xffffu) << 16);     // tile_dim0 = n8
    g1[4] = 0;                                         // tile_dim1/2 unused
    g1[5] = n8;                                        // tensor_dim0_stride
    g1[6] = 0;
    g1[7] = 0;
    i32x4 z4 = {0, 0, 0, 0};
#if defined(__clang_major__) && (__clang_major__ >= 23)
    i32x8 z8 = {0, 0, 0, 0, 0, 0, 0, 0};
    __builtin_amdgcn_tensor_load_to_lds(g0, g1, z4, z4, z8, 0);
#else
    __builtin_amdgcn_tensor_load_to_lds(g0, g1, z4, z4, 0);
#endif
    __builtin_amdgcn_s_wait_tensorcnt(0);
  }
  __syncthreads();
}

// ---------------- utility kernels ----------------
__global__ void fill_f32(float* p, float v, int n) {
  int i = blockIdx.x * blockDim.x + threadIdx.x;
  if (i < n) p[i] = v;
}
__global__ void fill_u32(unsigned* p, unsigned v, int n) {
  int i = blockIdx.x * blockDim.x + threadIdx.x;
  if (i < n) p[i] = v;
}
__global__ void axpy_add(float* p, const float* c, int n) {
  int i = blockIdx.x * blockDim.x + threadIdx.x;
  if (i < n) p[i] += c[i];
}
__global__ void init_h(float* h, const float* h0, int n) {
  int i = blockIdx.x * blockDim.x + threadIdx.x;
  if (i < n) h[i] = h0[i & (HH - 1)];
}

// Pack all weights to f16 in WMMA B-FRAGMENT-MAJOR order:
//   dest[((kt*NT + nt)*32 + lane)*16 + j] = W[kt*32 + (lane>>4)*16 + j][nt*16 + (lane&15)]
__global__ void pack_weights(const float* W_enc, const float* W_hop, const float* W_lt,
                             const float* W_tau, const float* W_g, const float* W_dec,
                             _Float16* enc16, _Float16* hop16, _Float16* lt16,
                             _Float16* tg16, _Float16* dec16) {
  int i = blockIdx.x * blockDim.x + threadIdx.x;
  int j = i & 15, lane = (i >> 4) & 31, tile = i >> 9;
  int kofs = ((lane >> 4) << 4) + j;   // half*16 + j
  int lr = lane & 15;
  if (i < 8192) {            // encoder: KT=4, NT=4, K=128, Nout=64
    int nt = tile & 3, kt = tile >> 2;
    enc16[i] = (_Float16)W_enc[(kt * 32 + kofs) * 64 + nt * 16 + lr];
  }
  if (i < 3 * 4096) {        // hop: 3 slices, KT=2, NT=4, 64x64 each
    int s = i / 4096, r = i % 4096;
    int jj = r & 15, ll = (r >> 4) & 31, tt = r >> 9;
    int nt = tt & 3, kt = tt >> 2;
    hop16[i] = (_Float16)W_hop[s * 4096 + (kt * 32 + ((ll >> 4) << 4) + jj) * 64 +
                               nt * 16 + (ll & 15)];
  }
  if (i < 4096) {            // larval: KT=2, NT=4
    int nt = tile & 3, kt = tile >> 2;
    lt16[i] = (_Float16)W_lt[(kt * 32 + kofs) * 64 + nt * 16 + lr];
  }
  if (i < 20480) {           // tau|g fused: KT=5, NT=8, K=133->160 pad, Nout=128
    int nt = tile % 8, kt = tile / 8;
    int k = kt * 32 + kofs, n = nt * 16 + lr;
    float v = 0.f;
    if (k < 133) v = (n < 64) ? W_tau[k * 64 + n] : W_g[k * 64 + (n - 64)];
    tg16[i] = (_Float16)v;
  }
  if (i < 8192) {            // decoder: KT=16, NT=1, K=512, Nout=3->16 pad
    int kt = tile;
    int k = kt * 32 + kofs, n = lr;
    dec16[i] = (_Float16)((n < 3) ? W_dec[k * 3 + n] : 0.f);
  }
}

// Edge gate MLP (4->16 tanh ->1 sigmoid) + degree count.
__global__ void gate_deg_kernel(const float* ea, const float* w1, const float* b1,
                                const float* w2, const float* b2, const int* dst,
                                float* gate, float* deg) {
  int e = blockIdx.x * blockDim.x + threadIdx.x;
  if (e >= EE) return;
  float a0 = ea[e * 4 + 0], a1 = ea[e * 4 + 1], a2 = ea[e * 4 + 2], a3 = ea[e * 4 + 3];
  float s = b2[0];
#pragma unroll
  for (int j = 0; j < 16; ++j) {
    float tv = a0 * w1[0 * 16 + j] + a1 * w1[1 * 16 + j] +
               a2 * w1[2 * 16 + j] + a3 * w1[3 * 16 + j] + b1[j];
    s += tanhf(tv) * w2[j];
  }
  gate[e] = fsigmoid(s);
  atomicAdd(&deg[dst[e]], 1.f);
}

// ---------------- WMMA GEMM kernels ----------------

__device__ __forceinline__ v16h load_bfrag(const _Float16* base, int tile, int lane) {
  return *(const v16h*)(base + (((size_t)tile * 32 + lane) << 4));
}
// A fragment from a contiguous f32 row: two aligned float4 pairs per half.
__device__ __forceinline__ v16h load_afrag_f32(const float* ar, int kt, int half, float scale) {
  const float4* p0 = (const float4*)(ar + kt * 32 + half * 16);
  const float4* p1 = (const float4*)(ar + kt * 32 + half * 16 + 8);
  float4 x0 = p0[0], x1 = p0[1], y0 = p1[0], y1 = p1[1];
  v16h a;
  a[0] = (_Float16)(x0.x * scale);  a[1] = (_Float16)(x0.y * scale);
  a[2] = (_Float16)(x0.z * scale);  a[3] = (_Float16)(x0.w * scale);
  a[4] = (_Float16)(x1.x * scale);  a[5] = (_Float16)(x1.y * scale);
  a[6] = (_Float16)(x1.z * scale);  a[7] = (_Float16)(x1.w * scale);
  a[8] = (_Float16)(y0.x * scale);  a[9] = (_Float16)(y0.y * scale);
  a[10] = (_Float16)(y0.z * scale); a[11] = (_Float16)(y0.w * scale);
  a[12] = (_Float16)(y1.x * scale); a[13] = (_Float16)(y1.y * scale);
  a[14] = (_Float16)(y1.z * scale); a[15] = (_Float16)(y1.w * scale);
  return a;
}

// u = fastkan_enc(x_t): A = rbf(x_t) on the fly (20000x128) @ W_enc (128x64)
// grid 1250, block 128 (4 waves = 4 N-tiles)
__global__ void encoder_gemm(const float* x, int t, const float* centers,
                             const _Float16* W16, const float* bias, float* u) {
  int wave = threadIdx.x >> 5, lane = threadIdx.x & 31;
  int half = lane >> 4, lr = lane & 15;
  int m0 = blockIdx.x * 16, n0 = wave * 16;
  // hoisted reciprocal: one v_rcp instead of per-element IEEE division
  float rden = (float)(NBASIS - 1) * frcp(centers[NBASIS - 1] - centers[0]);
  int m = m0 + lr, bb = m / NN, ii = m % NN;
  const float* xr = x + (((size_t)bb * TT + t) * NN + ii) * FF;
  v8f c = {};
#pragma unroll
  for (int kt = 0; kt < 4; ++kt) {
    v16h a;
#pragma unroll
    for (int j = 0; j < 16; ++j) {
      int k = kt * 32 + a_k(j, half);
      float d = (xr[k >> 3] - centers[k & 7]) * rden;
      a[j] = (_Float16)__expf(-d * d);
    }
    v16h bf = load_bfrag(W16, kt * 4 + wave, lane);
    c = __builtin_amdgcn_wmma_f32_16x16x32_f16(false, a, false, bf, (short)0, c, false, false);
  }
#pragma unroll
  for (int r = 0; r < 8; ++r) {
    int mm = m0 + half * 8 + r, n = n0 + lr;
    u[(size_t)mm * HH + n] = c[r] + bias[n];
  }
}

// hw = A(BN x 64) @ W(64x64); weights TDM-staged into LDS. grid 1250, block 128.
__global__ void hop_gemm(const float* A, const _Float16* W16f, float* C) {
  __shared__ __align__(32) _Float16 lw[4096];
  tdm_load_weights(lw, W16f, 4096 * 2 / 8);          // 8KB
  int wave = threadIdx.x >> 5, lane = threadIdx.x & 31;
  int half = lane >> 4, lr = lane & 15;
  int m0 = blockIdx.x * 16, n0 = wave * 16;
  const float* ar = A + (size_t)(m0 + lr) * HH;
  __builtin_prefetch(ar, 0, 3);
  v8f c = {};
#pragma unroll
  for (int kt = 0; kt < 2; ++kt) {
    v16h a = load_afrag_f32(ar, kt, half, 1.f);
    v16h bf = load_bfrag(lw, kt * 4 + wave, lane);   // ds_load path
    c = __builtin_amdgcn_wmma_f32_16x16x32_f16(false, a, false, bf, (short)0, c, false, false);
  }
#pragma unroll
  for (int r = 0; r < 8; ++r)
    C[(size_t)(m0 + half * 8 + r) * HH + n0 + lr] = c[r];
}

// P = P/3 + (agg/(deg+1)) @ W_lt. grid 1250, block 128.
__global__ void larval_gemm(const float* agg, const float* deg, const _Float16* W16f, float* P) {
  int wave = threadIdx.x >> 5, lane = threadIdx.x & 31;
  int half = lane >> 4, lr = lane & 15;
  int m0 = blockIdx.x * 16, n0 = wave * 16;
  const float* ar = agg + (size_t)(m0 + lr) * HH;
  float rd = frcp(deg[(m0 + lr) % NN] + 1.f);
  v8f c;
#pragma unroll
  for (int r = 0; r < 8; ++r)
    c[r] = P[(size_t)(m0 + half * 8 + r) * HH + n0 + lr] * (1.f / KHOP);
#pragma unroll
  for (int kt = 0; kt < 2; ++kt) {
    v16h a = load_afrag_f32(ar, kt, half, rd);
    v16h bf = load_bfrag(W16f, kt * 4 + wave, lane);
    c = __builtin_amdgcn_wmma_f32_16x16x32_f16(false, a, false, bf, (short)0, c, false, false);
  }
#pragma unroll
  for (int r = 0; r < 8; ++r)
    P[(size_t)(m0 + half * 8 + r) * HH + n0 + lr] = c[r];
}

// tg = z @ [W_tau|W_g], z=[h(64),env(5),p(64),pad] on the fly (K=160, Nout=128).
// Weights (40KB) TDM-staged into LDS. grid 1250, block 256 (8 waves).
__global__ void tg_gemm(const float* h, const float* p, const float* x, int t,
                        const _Float16* W16f, float* tg) {
  __shared__ __align__(32) _Float16 lw[20480];
  tdm_load_weights(lw, W16f, 20480 * 2 / 8);         // 40KB
  int wave = threadIdx.x >> 5, lane = threadIdx.x & 31;
  int half = lane >> 4, lr = lane & 15;
  int m0 = blockIdx.x * 16, n0 = wave * 16;
  int m = m0 + lr, bb = m / NN, ii = m % NN;
  const float* hr = h + (size_t)m * HH;
  const float* pr = p + (size_t)m * HH;
  const float* xr = x + (((size_t)bb * TT + t) * NN + ii) * FF;
  v8f c = {};
#pragma unroll
  for (int kt = 0; kt < 5; ++kt) {
    v16h a;
#pragma unroll
    for (int j = 0; j < 16; ++j) {
      int k = kt * 32 + a_k(j, half);
      float v;
      if (k < 64)       v = hr[k];
      else if (k < 69)  v = xr[8 + (k - 64)];   // TEMP,SAL,CU,CV,TR = feats 8..12
      else if (k < 133) v = pr[k - 69];
      else              v = 0.f;
      a[j] = (_Float16)v;
    }
    v16h bf = load_bfrag(lw, kt * 8 + wave, lane);
    c = __builtin_amdgcn_wmma_f32_16x16x32_f16(false, a, false, bf, (short)0, c, false, false);
  }
#pragma unroll
  for (int r = 0; r < 8; ++r)
    tg[(size_t)(m0 + half * 8 + r) * 128 + n0 + lr] = c[r];
}

// y = softplus(rbf(h).flat(512) @ W_dec(512x16 pad) + b). grid 1250, block 32.
__global__ void decoder_gemm(const float* h, const float* centers, const _Float16* W16,
                             const float* bias, float* out, int t) {
  int lane = threadIdx.x & 31;
  int half = lane >> 4, lr = lane & 15;
  int m0 = blockIdx.x * 16;
  float rden = (float)(NBASIS - 1) * frcp(centers[NBASIS - 1] - centers[0]);
  const float* hr = h + (size_t)(m0 + lr) * HH;
  v8f c = {};
#pragma unroll
  for (int kt = 0; kt < 16; ++kt) {
    v16h a;
#pragma unroll
    for (int j = 0; j < 16; ++j) {
      int k = kt * 32 + a_k(j, half);
      float d = (hr[k >> 3] - centers[k & 7]) * rden;
      a[j] = (_Float16)__expf(-d * d);
    }
    v16h bf = load_bfrag(W16, kt, lane);
    c = __builtin_amdgcn_wmma_f32_16x16x32_f16(false, a, false, bf, (short)0, c, false, false);
  }
  int n = lr;
  if (n < 3) {
#pragma unroll
    for (int r = 0; r < 8; ++r) {
      int m = m0 + half * 8 + r, bb = m / NN, ii = m % NN;
      float v = c[r] + bias[n];
      float sp = (v > 20.f) ? v : log1pf(__expf(v));
      out[(((size_t)bb * TT + t) * NN + ii) * 3 + n] = sp;
    }
  }
}

// ---------------- edge pipeline (f32 scatter path, L2-resident) ----------------

__global__ void node_dots(const float* hw, const float* asrc, const float* adst,
                          float* nds, float* ndd) {
  int m = blockIdx.x * blockDim.x + threadIdx.x;
  if (m >= BN) return;
  const float* r = hw + (size_t)m * HH;
  float s = 0.f, d = 0.f;
#pragma unroll 8
  for (int j = 0; j < HH; ++j) { s += r[j] * asrc[j]; d += r[j] * adst[j]; }
  nds[m] = s; ndd[m] = d;
}

__global__ void edge_score(const int* src, const int* dst, const float* nds,
                           const float* ndd, float* ebuf, unsigned* mmax) {
  int e = blockIdx.x * blockDim.x + threadIdx.x;
  int b = blockIdx.y;
  if (e >= EE) return;
  int si = src[e], di = dst[e];
  float v = nds[b * NN + si] + ndd[b * NN + di];
  v = (v > 0.f) ? v : 0.2f * v;                 // leaky_relu(0.2)
  ebuf[(size_t)b * EE + e] = v;
  atomicMax(&mmax[b * NN + di], fenc(v));
}

__global__ void edge_expsum(const int* dst, const float* ebuf,
                            const unsigned* mmax, float* dsum) {
  int e = blockIdx.x * blockDim.x + threadIdx.x;
  int b = blockIdx.y;
  if (e >= EE) return;
  int di = dst[e];
  atomicAdd(&dsum[b * NN + di], __expf(ebuf[(size_t)b * EE + e] - fdec(mmax[b * NN + di])));
}

// grid (EE/4, BB), block 256 (4 edges x 64 features)
__global__ void edge_aggr(const int* src, const int* dst, const float* ebuf,
                          const unsigned* mmax, const float* dsum,
                          const float* hw, float* outp) {
  int e = blockIdx.x * 4 + (threadIdx.x >> 6);
  int f = threadIdx.x & 63;
  int b = blockIdx.y;
  if (e >= EE) return;
  int si = src[e], di = dst[e];
  float w = __expf(ebuf[(size_t)b * EE + e] - fdec(mmax[b * NN + di])) *
            frcp(dsum[b * NN + di] + 1e-16f);
  atomicAdd(&outp[((size_t)b * NN + di) * HH + f], w * hw[((size_t)b * NN + si) * HH + f]);
}

__global__ void larval_aggr(const int* src, const int* dst, const float* gate,
                            const float* h, float* agg) {
  int e = blockIdx.x * 4 + (threadIdx.x >> 6);
  int f = threadIdx.x & 63;
  int b = blockIdx.y;
  if (e >= EE) return;
  atomicAdd(&agg[((size_t)b * NN + dst[e]) * HH + f],
            gate[e] * h[((size_t)b * NN + src[e]) * HH + f]);
}

// Liquid-cell update + LayerNorm + residual u_t. One wave per node, 2 feats/lane.
__global__ void ltc_update(const float* tg, const float* u, const float* b_tau,
                           const float* b_g, const float* gamma, const float* beta,
                           float* h) {
  int node = blockIdx.x * (blockDim.x >> 5) + (threadIdx.x >> 5);
  int lane = threadIdx.x & 31;
  if (node >= BN) return;
  const float dt = 1.f / TT;
  float hn[2];
#pragma unroll
  for (int q = 0; q < 2; ++q) {
    int j = lane + 32 * q;
    float hv = h[(size_t)node * HH + j];
    float tv = tg[(size_t)node * 128 + j] + b_tau[j];
    float gv = tanhf(tg[(size_t)node * 128 + 64 + j] + b_g[j]);
    float tau = 1.f + 9.f * fsigmoid(tv);          // TAU_MIN + (MAX-MIN)*sigmoid
    hn[q] = hv + dt * (gv - hv) * frcp(tau);
  }
  float s = hn[0] + hn[1], s2 = hn[0] * hn[0] + hn[1] * hn[1];
#pragma unroll
  for (int off = 16; off; off >>= 1) {
    s  += __shfl_xor(s, off, 32);
    s2 += __shfl_xor(s2, off, 32);
  }
  float mu = s / 64.f;
  float inv = rsqrtf(s2 / 64.f - mu * mu + 1e-5f);
#pragma unroll
  for (int q = 0; q < 2; ++q) {
    int j = lane + 32 * q;
    h[(size_t)node * HH + j] = (hn[q] - mu) * inv * gamma[j] + beta[j] +
                               u[(size_t)node * HH + j];
  }
}

// ---------------- host orchestration ----------------
extern "C" void kernel_launch(void* const* d_in, const int* in_sizes, int n_in,
                              void* d_out, int out_size, void* d_ws, size_t ws_size,
                              hipStream_t stream) {
  (void)in_sizes; (void)n_in; (void)out_size; (void)ws_size;
  const float* x         = (const float*)d_in[0];
  const float* edge_attr = (const float*)d_in[1];
  const float* c_enc     = (const float*)d_in[2];
  const float* W_enc     = (const float*)d_in[3];
  const float* b_enc     = (const float*)d_in[4];
  const float* W_hop     = (const float*)d_in[5];
  const float* a_src     = (const float*)d_in[6];
  const float* a_dst     = (const float*)d_in[7];
  const float* w_lt1     = (const float*)d_in[8];
  const float* b_lt1     = (const float*)d_in[9];
  const float* w_lt2     = (const float*)d_in[10];
  const float* b_lt2     = (const float*)d_in[11];
  const float* W_lt      = (const float*)d_in[12];
  const float* W_tau     = (const float*)d_in[13];
  const float* b_tau     = (const float*)d_in[14];
  const float* W_g       = (const float*)d_in[15];
  const float* b_g       = (const float*)d_in[16];
  const float* gamma     = (const float*)d_in[17];
  const float* beta      = (const float*)d_in[18];
  const float* c_dec     = (const float*)d_in[19];
  const float* W_dec     = (const float*)d_in[20];
  const float* b_dec     = (const float*)d_in[21];
  const float* h0        = (const float*)d_in[22];
  const int*   eidx      = (const int*)d_in[23];
  const int* src = eidx;
  const int* dst = eidx + EE;
  float* out = (float*)d_out;

  float* ws = (float*)d_ws;
  size_t off = 0;
  auto alloc = [&](size_t n) { float* q = ws + off; off += n; return q; };
  float* h    = alloc((size_t)BN * HH);
  float* u    = alloc((size_t)BN * HH);
  float* p    = alloc((size_t)BN * HH);
  float* cur  = alloc((size_t)BN * HH);
  float* nxt  = alloc((size_t)BN * HH);
  float* hw   = alloc((size_t)BN * HH);
  float* agg  = alloc((size_t)BN * HH);
  float* tg   = alloc((size_t)BN * 128);
  float* gate = alloc(EE);
  float* deg  = alloc(NN);
  float* ebuf = alloc((size_t)BB * EE);
  unsigned* mmax = (unsigned*)alloc(BB * NN);
  float* dsum = alloc(BB * NN);
  float* nds  = alloc(BB * NN);
  float* ndd  = alloc(BB * NN);
  _Float16* enc16 = (_Float16*)alloc(128 * 64 / 2);
  _Float16* hop16 = (_Float16*)alloc(3 * 64 * 64 / 2);
  _Float16* lt16  = (_Float16*)alloc(64 * 64 / 2);
  _Float16* tg16  = (_Float16*)alloc(160 * 128 / 2);
  _Float16* dec16 = (_Float16*)alloc(512 * 16 / 2);

  dim3 thr(256);
  auto nb = [](long n, int b) { return dim3((unsigned)((n + b - 1) / b)); };

  pack_weights<<<nb(20480, 256), thr, 0, stream>>>(W_enc, W_hop, W_lt, W_tau, W_g,
                                                   W_dec, enc16, hop16, lt16, tg16, dec16);
  fill_f32<<<nb(NN, 256), thr, 0, stream>>>(deg, 0.f, NN);
  gate_deg_kernel<<<nb(EE, 256), thr, 0, stream>>>(edge_attr, w_lt1, b_lt1, w_lt2,
                                                   b_lt2, dst, gate, deg);
  init_h<<<nb((size_t)BN * HH, 256), thr, 0, stream>>>(h, h0, BN * HH);

  for (int t = 0; t < TT; ++t) {
    encoder_gemm<<<dim3(1250), dim3(128), 0, stream>>>(x, t, c_enc, enc16, b_enc, u);
    fill_f32<<<nb((size_t)BN * HH, 256), thr, 0, stream>>>(p, 0.f, BN * HH);

    const float* curp = h;
    float* outA = cur; float* outB = nxt;
    for (int k = 0; k < KHOP; ++k) {
      hop_gemm<<<dim3(1250), dim3(128), 0, stream>>>(curp, hop16 + (size_t)k * 4096, hw);
      node_dots<<<nb(BN, 256), thr, 0, stream>>>(hw, a_src + k * HH, a_dst + k * HH, nds, ndd);
      fill_u32<<<nb(BB * NN, 256), thr, 0, stream>>>(mmax, ENC_NEG_INF, BB * NN);
      fill_f32<<<nb(BB * NN, 256), thr, 0, stream>>>(dsum, 0.f, BB * NN);
      edge_score<<<dim3((EE + 255) / 256, BB), thr, 0, stream>>>(src, dst, nds, ndd, ebuf, mmax);
      edge_expsum<<<dim3((EE + 255) / 256, BB), thr, 0, stream>>>(dst, ebuf, mmax, dsum);
      fill_f32<<<nb((size_t)BN * HH, 256), thr, 0, stream>>>(outA, 0.f, BN * HH);
      edge_aggr<<<dim3(EE / 4, BB), thr, 0, stream>>>(src, dst, ebuf, mmax, dsum, hw, outA);
      axpy_add<<<nb((size_t)BN * HH, 256), thr, 0, stream>>>(p, outA, BN * HH);
      curp = outA; float* tsw = outA; outA = outB; outB = tsw;
    }
    // larval transport
    fill_f32<<<nb((size_t)BN * HH, 256), thr, 0, stream>>>(agg, 0.f, BN * HH);
    larval_aggr<<<dim3(EE / 4, BB), thr, 0, stream>>>(src, dst, gate, h, agg);
    larval_gemm<<<dim3(1250), dim3(128), 0, stream>>>(agg, deg, lt16, p);
    // liquid cell gates
    tg_gemm<<<dim3(1250), dim3(256), 0, stream>>>(h, p, x, t, tg16, tg);
    ltc_update<<<dim3(BN / 8), thr, 0, stream>>>(tg, u, b_tau, b_g, gamma, beta, h);
    // decode
    decoder_gemm<<<dim3(1250), dim3(32), 0, stream>>>(h, c_dec, dec16, b_dec, out, t);
  }
}
